// BiRNN_15607911153798
// MI455X (gfx1250) — matrix operand order repaired and checked by reading, not compiled, not run on hardware
//
#include <hip/hip_runtime.h>
#include <hip/hip_bf16.h>

// BiRNN (tanh) with per-step standardnorm recurrent state.
// B=64, T=1024, E=512, H=512.
//
// Phase 0: convert Wih_f/Whh_f/Wih_b/Whh_b to f16 in workspace (2 MB).
// Phase 1: xp = emb @ Wih^T + bih for both directions, written directly into
//          d_out's hiddens region ([B,T,2H]): fwd -> cols [0,H), bwd -> [H,2H).
//          Each wave computes a 16x64 output strip (4 N-tiles sharing one A
//          fragment) for 4x A reuse.
// Phase 2: 8 persistent workgroups (2 dirs x 4 batch-quarters of 16 rows),
//          1024 threads = 32 waves each; wave w owns cols [16w,16w+16).
//          Sequential over T with v_wmma_f32_16x16x32_f16 on LDS-resident
//          normalized state; stdnorm via shfl_xor + LDS f32 atomics.

typedef __attribute__((ext_vector_type(16))) _Float16 v16h;
typedef __attribute__((ext_vector_type(8)))  _Float16 v8h;
typedef __attribute__((ext_vector_type(8)))  float    v8f;

#define BB    64
#define TT    1024
#define EE    512
#define HH    512
#define TWOH  1024
#define WMAT  (HH * EE)   // 262144 elements per weight matrix

// ---------------- Phase 0: f32 -> f16 weight conversion ----------------
__global__ void cvt_weights_kernel(const float* __restrict__ Wih_f,
                                   const float* __restrict__ Whh_f,
                                   const float* __restrict__ Wih_b,
                                   const float* __restrict__ Whh_b,
                                   _Float16* __restrict__ dst) {
    int i = blockIdx.x * blockDim.x + threadIdx.x;   // 0 .. 4*WMAT-1
    int m = i >> 18;                                  // which matrix
    int e = i & (WMAT - 1);
    const float* src = (m == 0) ? Wih_f : (m == 1) ? Whh_f : (m == 2) ? Wih_b : Whh_b;
    dst[i] = (_Float16)src[e];
}

// ---------------- Phase 1: input projection (WMMA GEMM) ----------------
// One wave per 16x64 output strip: 4 N-tiles share each A fragment.
// Strip space: 4096 row-tiles x 8 col-groups x 2 dirs = 65536 waves.
__global__ void proj_kernel(const float* __restrict__ emb,
                            const _Float16* __restrict__ Wih16_f,
                            const _Float16* __restrict__ Wih16_b,
                            const float* __restrict__ bih_f,
                            const float* __restrict__ bih_b,
                            float* __restrict__ hiddens) {
    int wave = (blockIdx.x * blockDim.x + threadIdx.x) >> 5;
    int lane = threadIdx.x & 31;
    int dir     = wave >> 15;         // 32768 strips per direction
    int rem     = wave & 32767;
    int rowTile = rem >> 3;           // 4096 row tiles over B*T
    int colGrp  = rem & 7;            // 8 col groups of 64 over H

    const _Float16* W    = dir ? Wih16_b : Wih16_f;
    const float*    bias = dir ? bih_b   : bih_f;

    int r    = rowTile * 16 + (lane & 15);   // A row (M = lane%16 for both halves)
    int nB   = colGrp * 64 + (lane & 15);    // first B/C column of the strip
    int hiK  = lane >> 4;                    // lane half selects K sub-block
    int kbAo = hiK * 16;                     // A K offsets {0..15} / {16..31}
    int kbBo = hiK * 16;                     // B K offsets {0..15} / {16..31}

    v8f acc0 = {}, acc1 = {}, acc2 = {}, acc3 = {};
    const float* arow = emb + (long)r * EE;
    // B[k][n] = Wih[n][k]: column n of B is row n of Wih (contiguous in k)
    const _Float16* bc0 = W + (long)(nB +  0) * EE + kbBo;
    const _Float16* bc1 = W + (long)(nB + 16) * EE + kbBo;
    const _Float16* bc2 = W + (long)(nB + 32) * EE + kbBo;
    const _Float16* bc3 = W + (long)(nB + 48) * EE + kbBo;
#pragma unroll 4
    for (int kt = 0; kt < 16; ++kt) {
        const float* ap = arow + kt * 32 + kbAo;
        v16h a;
#pragma unroll
        for (int j = 0; j < 8; ++j) {
            a[j]     = (_Float16)ap[j];
            a[j + 8] = (_Float16)ap[8 + j];
        }
        v16h b0 = *(const v16h*)(bc0 + kt * 32);
        v16h b1 = *(const v16h*)(bc1 + kt * 32);
        v16h b2 = *(const v16h*)(bc2 + kt * 32);
        v16h b3 = *(const v16h*)(bc3 + kt * 32);
        acc0 = __builtin_amdgcn_wmma_f32_16x16x32_f16(false, a, false, b0,
                                                      (short)0, acc0, false, false);
        acc1 = __builtin_amdgcn_wmma_f32_16x16x32_f16(false, a, false, b1,
                                                      (short)0, acc1, false, false);
        acc2 = __builtin_amdgcn_wmma_f32_16x16x32_f16(false, a, false, b2,
                                                      (short)0, acc2, false, false);
        acc3 = __builtin_amdgcn_wmma_f32_16x16x32_f16(false, a, false, b3,
                                                      (short)0, acc3, false, false);
    }
    int m0 = hiK * 8;
#pragma unroll
    for (int c = 0; c < 4; ++c) {
        const v8f acc = (c == 0) ? acc0 : (c == 1) ? acc1 : (c == 2) ? acc2 : acc3;
        int   n  = nB + c * 16;
        float bv = bias[n];
#pragma unroll
        for (int v = 0; v < 8; ++v) {
            int gr = rowTile * 16 + m0 + v;      // flattened b*T + t index
            hiddens[(long)gr * TWOH + dir * HH + n] = acc[v] + bv;
        }
    }
}

// ---------------- Phase 2: sequential recurrence ----------------
// grid = 8: blockIdx.x -> {dir = x>>2, batch quarter = x&3 (16 rows)}.
__global__ __launch_bounds__(1024) void rnn_rec_kernel(
        float* __restrict__ hiddens,       // [B,T,2H]; holds xp on entry
        float* __restrict__ last,          // [B,2H]
        const _Float16* __restrict__ Whh16_f,
        const _Float16* __restrict__ Whh16_b,
        const float* __restrict__ bhh_f,
        const float* __restrict__ bhh_b) {
    int dir   = blockIdx.x >> 2;
    int bBase = (blockIdx.x & 3) * 16;
    const _Float16* W    = dir ? Whh16_b : Whh16_f;
    const float*    bias = dir ? bhh_b   : bhh_f;
    int dirOff = dir * HH;

    int wave = threadIdx.x >> 5;
    int lane = threadIdx.x & 31;
    int nLoc = wave * 16 + (lane & 15);   // column in [0,H)
    int m0   = (lane >> 4) * 8;           // C rows per lane: m0..m0+7
    int rA   = lane & 15;                 // A row
    int kbAo = (lane < 16) ? 0 : 16;
    int kbBo = (lane < 16) ? 0 : 16;

    __shared__ _Float16 hn[16 * HH];      // normalized state (A operand), 16 KB
    __shared__ float rsum[16], rsq[16];

    for (int i = threadIdx.x; i < 16 * HH; i += 1024) hn[i] = (_Float16)0.f;
    __syncthreads();                       // h0 = zeros

    float bv = bias[nLoc];
    const _Float16* bcol = W + (long)nLoc * HH + kbBo;   // Whh^T col n = Whh row n

    for (int step = 0; step < TT; ++step) {
        int t = dir ? (TT - 1 - step) : step;

        if (threadIdx.x < 16) { rsum[threadIdx.x] = 0.f; rsq[threadIdx.x] = 0.f; }

        // --- acc = stdnorm(h_prev) @ Whh^T, K = 512 ---
        v8f acc = {};
#pragma unroll 4
        for (int kt = 0; kt < 16; ++kt) {
            const v8h* hp = (const v8h*)&hn[rA * HH + kt * 32 + kbAo];
            v8h lo = hp[0];
            v8h hi = hp[1];                // +8 halves
            v16h a = __builtin_shufflevector(lo, hi, 0, 1, 2, 3, 4, 5, 6, 7,
                                                     8, 9, 10, 11, 12, 13, 14, 15);
            v16h b = *(const v16h*)(bcol + kt * 32);
            acc = __builtin_amdgcn_wmma_f32_16x16x32_f16(false, a, false, b,
                                                         (short)0, acc, false, false);
        }

        // --- h = tanh(xp + acc + bhh); overwrite xp with pre-norm h ---
        float h[8], sv[8], qv[8];
#pragma unroll
        for (int v = 0; v < 8; ++v) {
            int brow = bBase + m0 + v;
            float* p = &hiddens[((long)brow * TT + t) * TWOH + dirOff + nLoc];
            float hv = tanhf(acc[v] + *p + bv);
            *p = hv;
            h[v] = hv;
            sv[v] = hv;
            qv[v] = hv * hv;
        }

        // --- row sums over this wave's 16 columns (per 16-lane half) ---
#pragma unroll
        for (int off = 1; off < 16; off <<= 1) {
#pragma unroll
            for (int v = 0; v < 8; ++v) {
                sv[v] += __shfl_xor(sv[v], off, 32);
                qv[v] += __shfl_xor(qv[v], off, 32);
            }
        }

        __syncthreads();                   // rsum/rsq zeroed, all waves here
        if ((lane & 15) == 0) {
#pragma unroll
            for (int v = 0; v < 8; ++v) {
                atomicAdd(&rsum[m0 + v], sv[v]);
                atomicAdd(&rsq[m0 + v], qv[v]);
            }
        }
        __syncthreads();                   // reductions complete

        // --- stdnorm (unbiased, ddof=1): hn = (h - mean) / (1e-10 + std) ---
#pragma unroll
        for (int v = 0; v < 8; ++v) {
            int   rr   = m0 + v;
            float sum  = rsum[rr];
            float mean = sum * (1.0f / 512.0f);
            float var  = (rsq[rr] - sum * mean) * (1.0f / 511.0f);
            float sd   = sqrtf(var > 0.f ? var : 0.f);
            float hv   = (h[v] - mean) / (1e-10f + sd);
            hn[rr * HH + nLoc] = (_Float16)hv;
            if (step == TT - 1)
                last[(long)(bBase + rr) * TWOH + dirOff + nLoc] = hv;
        }
        __syncthreads();                   // hn ready for next step
    }
}

extern "C" void kernel_launch(void* const* d_in, const int* in_sizes, int n_in,
                              void* d_out, int out_size, void* d_ws, size_t ws_size,
                              hipStream_t stream) {
    const float* x     = (const float*)d_in[0];
    const float* Wih_f = (const float*)d_in[1];
    const float* Whh_f = (const float*)d_in[2];
    const float* bih_f = (const float*)d_in[3];
    const float* bhh_f = (const float*)d_in[4];
    const float* Wih_b = (const float*)d_in[5];
    const float* Whh_b = (const float*)d_in[6];
    const float* bih_b = (const float*)d_in[7];
    const float* bhh_b = (const float*)d_in[8];

    float* out  = (float*)d_out;
    float* last = out;                       // [64, 1024]
    float* hid  = out + (long)BB * TWOH;     // [64, 1024, 1024]

    _Float16* w16     = (_Float16*)d_ws;     // 4 x 512KB f16 weights
    _Float16* Wih16_f = w16;
    _Float16* Whh16_f = w16 + WMAT;
    _Float16* Wih16_b = w16 + 2 * WMAT;
    _Float16* Whh16_b = w16 + 3 * WMAT;

    // Phase 0: weight conversion (4*262144 elems / 256)
    cvt_weights_kernel<<<4096, 256, 0, stream>>>(Wih_f, Whh_f, Wih_b, Whh_b, w16);

    // Phase 1: 65536 strips (16x64 each), 8 waves per 256-thread block
    proj_kernel<<<8192, 256, 0, stream>>>(x, Wih16_f, Wih16_b, bih_f, bih_b, hid);

    // Phase 2: 2 dirs x 4 batch quarters, 1024 threads each
    rnn_rec_kernel<<<8, 1024, 0, stream>>>(hid, last, Whh16_f, Whh16_b, bhh_f, bhh_b);
}
